// EdgeConv2dDiff_85804856640068
// MI455X (gfx1250) — matrix operand order repaired and checked by reading, not compiled, not run on hardware
//
#include <hip/hip_runtime.h>
#include <hip/hip_bf16.h>
#include <stdint.h>

// ---------------------------------------------------------------------------
// EdgeConv (concat(x_i, x_j - x_i) -> 1x1 conv -> ReLU -> max over K)
// GEMM formulation: C[64 x 800000] = W[64 x 128] * E[128 x 800000]
// bf16 WMMA (V_WMMA_F32_16X16X32_BF16), f32 accumulation -> HBM-bound.
// ---------------------------------------------------------------------------

typedef __attribute__((ext_vector_type(16))) __bf16 v16bf;
typedef __attribute__((ext_vector_type(8)))  float  v8f;

#define CIN    64
#define CC     128          // 2*C  (GEMM K dim)
#define NCOUT  64
#define KNB    16           // neighbors per point
#define TILE_N 8            // points per workgroup
#define COLS   128          // TILE_N * KNB positions per workgroup
#define LROW   68           // padded row stride in dwords (64 kpairs + 4 pad)

// LDS layout (u32 units):
//   Wl [64 rows][LROW]  : W as packed bf16 K-pairs   (4352 u32)
//   El [128 cols][LROW] : E column-major, packed bf16 K-pairs (8704 u32)
#define WL_U32  (NCOUT * LROW)
#define EL_U32  (COLS  * LROW)
#define LDS_U32 (WL_U32 + EL_U32)     // 13056 u32 = 52224 B

__device__ __forceinline__ uint32_t pack_bf16(float a, float b) {
    // round-to-nearest-even f32 -> bf16, packed (a in low 16, b in high 16)
    uint32_t ua = __float_as_uint(a), ub = __float_as_uint(b);
    ua += 0x7FFFu + ((ua >> 16) & 1u);
    ub += 0x7FFFu + ((ub >> 16) & 1u);
    return (ua >> 16) | (ub & 0xFFFF0000u);
}

union FragU { struct { uint4 lo, hi; } q; v16bf v; };

__global__ __launch_bounds__(256)
void edgeconv_wmma_kernel(const float* __restrict__ xi,
                          const float* __restrict__ xj,
                          const float* __restrict__ Wg,
                          const float* __restrict__ bias,
                          float* __restrict__ out, int Npts) {
    __shared__ uint32_t lds[LDS_U32];
    uint32_t* Wl = lds;
    uint32_t* El = lds + WL_U32;

    const int t    = threadIdx.x;
    const int lane = t & 31;
    const int wave = t >> 5;        // 0..7, one 16-col tile each
    const int r    = lane & 15;     // row/col within WMMA tile
    const int h    = lane >> 4;     // lane half

    const int    n0    = blockIdx.x * TILE_N;
    const size_t pbase = (size_t)n0 * KNB;       // first flat position

    // ---- Stage W: f32 [64][128] -> packed bf16 K-pairs [64][LROW] ----
    {
        const int row = t >> 2;                  // 0..63
        const int j0  = (t & 3) * 16;            // kpair start
        const float* wrow = Wg + row * CC;
        uint32_t* wl = Wl + row * LROW;
        #pragma unroll 4
        for (int j = 0; j < 16; ++j) {
            const int kp = j0 + j;
            wl[kp] = pack_bf16(wrow[2 * kp], wrow[2 * kp + 1]);
        }
    }

    // ---- Stage E: col-major packed bf16 K-pairs.
    //      rows 0..63  = x_i ; rows 64..127 = x_j - x_i ----
    {
        const int col = t & 127;                 // position within tile
        const int jh  = (t >> 7) * 16;           // each half-block does 16 kpairs
        const size_t p  = pbase + (size_t)col;
        const size_t cs = (size_t)Npts * KNB;    // channel stride in x
        const float* xip = xi + p;
        const float* xjp = xj + p;
        uint32_t* ecol = El + col * LROW;
        for (int jj = jh; jj < jh + 16; jj += 4) {
            uint4 u, v;
            uint32_t* up = &u.x; uint32_t* vp = &v.x;
            #pragma unroll
            for (int q = 0; q < 4; ++q) {
                const int j = jj + q;
                const float a0 = xip[(size_t)(2 * j)     * cs];
                const float a1 = xip[(size_t)(2 * j + 1) * cs];
                const float b0 = xjp[(size_t)(2 * j)     * cs];
                const float b1 = xjp[(size_t)(2 * j + 1) * cs];
                up[q] = pack_bf16(a0, a1);
                vp[q] = pack_bf16(b0 - a0, b1 - a1);
            }
            *(uint4*)(ecol + jj)      = u;   // x_i pairs      (kpairs 0..31)
            *(uint4*)(ecol + 32 + jj) = v;   // (x_j-x_i) pairs (kpairs 32..63)
        }
    }

    __syncthreads();

    // ---- GEMM: 4 row tiles (couts) x 1 col tile (this wave's point) ----
    v8f acc[4];
    #pragma unroll
    for (int m = 0; m < 4; ++m) { v8f z = {}; acc[m] = z; }

    #pragma unroll
    for (int kk = 0; kk < 4; ++kk) {             // K = kk*32 .. kk*32+31
        // A fragments: 16x32 bf16, lane r = row, VGPR kpairs per ISA layout
        FragU a[4];
        #pragma unroll
        for (int m = 0; m < 4; ++m) {
            const uint32_t* ab = Wl + (m * 16 + r) * LROW + kk * 16;
            a[m].q.lo = *(const uint4*)(ab + h * 4);        // K  h*8 .. h*8+7
            a[m].q.hi = *(const uint4*)(ab + 8 + h * 4);    // K 16+h*8 .. +7
        }
        // B fragment: 32x16 bf16, lane r = col, kpairs h*4 .. h*4+3 and 8+h*4 .. 8+h*4+3
        FragU bf;
        const uint32_t* bb = El + (wave * 16 + r) * LROW + kk * 16;
        bf.q.lo = *(const uint4*)(bb + h * 4);
        bf.q.hi = *(const uint4*)(bb + 8 + h * 4);

        #pragma unroll
        for (int m = 0; m < 4; ++m) {
            acc[m] = __builtin_amdgcn_wmma_f32_16x16x32_bf16(
                false, a[m].v, false, bf.v, (short)0, acc[m], false, false);
        }
    }

    __syncthreads();                 // done with Wl; reuse as result staging
    float* resLds = reinterpret_cast<float*>(lds);   // [64 couts][8 points]

    // ---- Epilogue: bias + ReLU, max over K (= 16 D-tile columns) ----
    const int nl = wave;             // this wave's point within tile
    #pragma unroll
    for (int m = 0; m < 4; ++m) {
        #pragma unroll
        for (int d = 0; d < 8; ++d) {
            const int cout = m * 16 + h * 8 + d;   // D layout: lanes 0-15 M=d, 16-31 M=d+8
            float v = acc[m][d] + bias[cout];
            v = fmaxf(v, 0.0f);
            // max over the 16 lanes of this half (the point's 16 neighbors)
            v = fmaxf(v, __shfl_xor(v, 1, 32));
            v = fmaxf(v, __shfl_xor(v, 2, 32));
            v = fmaxf(v, __shfl_xor(v, 4, 32));
            v = fmaxf(v, __shfl_xor(v, 8, 32));
            if (r == 0) resLds[cout * TILE_N + nl] = v;
        }
    }
    __syncthreads();

    // ---- Coalesced output: out[cout][n0 .. n0+7] ----
    {
        const int cout = t >> 2;                  // 0..63
        const int ib   = (t & 3) * 2;             // 0,2,4,6
        float2 vv = *(const float2*)(resLds + cout * TILE_N + ib);
        *(float2*)(out + (size_t)cout * Npts + n0 + ib) = vv;
    }
}

extern "C" void kernel_launch(void* const* d_in, const int* in_sizes, int n_in,
                              void* d_out, int out_size, void* d_ws, size_t ws_size,
                              hipStream_t stream) {
    const float* xi   = (const float*)d_in[0];
    const float* xj   = (const float*)d_in[1];
    const float* Wg   = (const float*)d_in[2];
    const float* bias = (const float*)d_in[3];
    float* out = (float*)d_out;
    const int Npts = 50000;                       // N; N % TILE_N == 0
    dim3 grid(Npts / TILE_N), block(256);
    edgeconv_wmma_kernel<<<grid, block, 0, stream>>>(xi, xj, Wg, bias, out, Npts);
}
